// PointNet_v2_21586505630006
// MI455X (gfx1250) — compile-verified, matrix operand order checked by reading, not run on hardware
//
// PointNet++ forward for MI455X (gfx1250, wave32, WMMA).
// GEMMs run v_wmma_f32_16x16x32_f16 with f32 accumulate; each wave computes a
// 16x64 C strip (4 accumulators, A-fragment reuse); all 5 fragments per K-step
// are loaded into distinct registers so loads clause together ahead of the
// 4 WMMAs. FPS / ball-query stage segment xyz into LDS with gfx1250 async
// LDS loads (ASYNCcnt).
#include <hip/hip_runtime.h>

typedef __attribute__((ext_vector_type(16))) _Float16 v16h;
typedef __attribute__((ext_vector_type(8)))  float    v8f;
typedef __attribute__((ext_vector_type(4)))  unsigned int u32x4;

#define EPSN 1e-5f
#define NT   32768
#define GSEG 32

#if __has_builtin(__builtin_amdgcn_global_load_async_to_lds_b32) && \
    __has_builtin(__builtin_amdgcn_s_wait_asynccnt)
#define HAVE_ASYNC_LDS 1
#else
#define HAVE_ASYNC_LDS 0
#endif

// builtin signature (from hipcc diagnostic): (global int*, local int*, Ii, Ii)
#define AS1I(p) ((__attribute__((address_space(1))) int*)(p))
#define AS3I(p) ((__attribute__((address_space(3))) int*)(p))

// ---------------------------------------------------------------- reductions
static __device__ __forceinline__ float blockReduceSum256(float v, float* sh) {
  int t = threadIdx.x;
  sh[t] = v; __syncthreads();
  for (int off = 128; off > 0; off >>= 1) {
    if (t < off) sh[t] += sh[t + off];
    __syncthreads();
  }
  float r = sh[0]; __syncthreads();
  return r;
}

// ---------------------------------------------------------------- WMMA GEMM
// C[M,N] = act(A[M,Kp] * B[N,Kp]^T + bias).  A,B f16 row-major (K contiguous),
// Kp multiple of 32, M multiple of 16, N multiple of 64 in this network.
// Block = 8 waves = 8 M-tiles; each wave computes a 16x64 strip (4 N-tiles).
__global__ __launch_bounds__(256)
void k_wmma_gemm(const _Float16* __restrict__ A, const _Float16* __restrict__ B,
                 float* __restrict__ C, const float* __restrict__ bias,
                 int M, int N, int Kp, int ldc, int relu) {
  const int lane  = threadIdx.x & 31;
  const int wave  = threadIdx.x >> 5;
  const int tileM = blockIdx.x * 8 + wave;
  if (tileM * 16 >= M) return;                    // wave-uniform exit
  const int n0    = blockIdx.y * 64;
  const int half  = lane >> 4;                    // K-half selector
  const int l16   = lane & 15;                    // M (for A) / N (for B) index
  const _Float16* Ap = A + (size_t)(tileM * 16 + l16) * Kp;
  const _Float16* Bp0 = B + (size_t)(n0 +      l16) * Kp;
  const _Float16* Bp1 = B + (size_t)(n0 + 16 + l16) * Kp;
  const _Float16* Bp2 = B + (size_t)(n0 + 32 + l16) * Kp;
  const _Float16* Bp3 = B + (size_t)(n0 + 48 + l16) * Kp;
  v8f acc0 = (v8f){0.f, 0.f, 0.f, 0.f, 0.f, 0.f, 0.f, 0.f};
  v8f acc1 = acc0, acc2 = acc0, acc3 = acc0;
  for (int k0 = 0; k0 < Kp; k0 += 32) {
    const int ko = k0 + 8 * half;
    // halves 0..7  <- K = k0 + 8*half + 0..7   (one b128 load)
    // halves 8..15 <- K = k0 + 16 + 8*half + 0..7
    union F { u32x4 q[2]; v16h h; } fa, fb0, fb1, fb2, fb3;
    fa.q[0]  = *(const u32x4*)(Ap + ko);
    fa.q[1]  = *(const u32x4*)(Ap + ko + 16);
    fb0.q[0] = *(const u32x4*)(Bp0 + ko);
    fb0.q[1] = *(const u32x4*)(Bp0 + ko + 16);
    fb1.q[0] = *(const u32x4*)(Bp1 + ko);
    fb1.q[1] = *(const u32x4*)(Bp1 + ko + 16);
    fb2.q[0] = *(const u32x4*)(Bp2 + ko);
    fb2.q[1] = *(const u32x4*)(Bp2 + ko + 16);
    fb3.q[0] = *(const u32x4*)(Bp3 + ko);
    fb3.q[1] = *(const u32x4*)(Bp3 + ko + 16);
    if (k0 + 32 < Kp) {                           // global_prefetch_b8 next tile
      __builtin_prefetch(Ap + k0 + 32, 0, 1);
      __builtin_prefetch(Bp0 + k0 + 32, 0, 1);
      __builtin_prefetch(Bp2 + k0 + 32, 0, 1);
    }
    acc0 = __builtin_amdgcn_wmma_f32_16x16x32_f16(
        false, fa.h, false, fb0.h, (short)0, acc0, false, false);
    acc1 = __builtin_amdgcn_wmma_f32_16x16x32_f16(
        false, fa.h, false, fb1.h, (short)0, acc1, false, false);
    acc2 = __builtin_amdgcn_wmma_f32_16x16x32_f16(
        false, fa.h, false, fb2.h, (short)0, acc2, false, false);
    acc3 = __builtin_amdgcn_wmma_f32_16x16x32_f16(
        false, fa.h, false, fb3.h, (short)0, acc3, false, false);
  }
  v8f accs[4] = {acc0, acc1, acc2, acc3};
#pragma unroll
  for (int t = 0; t < 4; ++t) {                   // after all WMMAs (EXEC rule)
    const int n = n0 + 16 * t + l16;
    if (n >= N) continue;
    const float bv = bias ? bias[n] : 0.f;
#pragma unroll
    for (int j = 0; j < 8; ++j) {                 // elem j: M = j + 8*half
      int m = tileM * 16 + half * 8 + j;
      float v = accs[t][j] + bv;
      if (relu) v = fmaxf(v, 0.f);
      C[(size_t)m * ldc + n] = v;
    }
  }
}

// ------------------------------------------------------------- small kernels
__global__ void k_fill_h16(_Float16* p, long n) {
  for (long i = (long)blockIdx.x * blockDim.x + threadIdx.x; i < n;
       i += (long)gridDim.x * blockDim.x)
    p[i] = (_Float16)0.f;
}

// dst[r,c] (rowsP x colsP f16) = (r<rows && c<cols) ? Mat[r][c] : 0
// trans==0: Mat[r][c]=src[r*srcld+c]; trans==1: Mat[r][c]=src[c*srcld+r]
__global__ void k_pad16(const float* __restrict__ src, _Float16* __restrict__ dst,
                        int rows, int cols, int colsP, long total, int trans,
                        int srcld) {
  long i = (long)blockIdx.x * blockDim.x + threadIdx.x;
  if (i >= total) return;
  int r = (int)(i / colsP), c = (int)(i % colsP);
  float v = 0.f;
  if (r < rows && c < cols)
    v = trans ? src[(size_t)c * srcld + r] : src[(size_t)r * srcld + c];
  dst[i] = (_Float16)v;
}

__global__ void k_pts_transpose(const float* __restrict__ s, float* __restrict__ d,
                                int n) {   // [3,N] -> [N,3]
  int i = blockIdx.x * blockDim.x + threadIdx.x;
  if (i >= n) return;
  d[i * 3 + 0] = s[i];
  d[i * 3 + 1] = s[n + i];
  d[i * 3 + 2] = s[2 * n + i];
}

__global__ void k_apply3(const float* __restrict__ x, const float* __restrict__ T,
                         float* __restrict__ y, int n) {  // y = x[N,3] @ T[3,3]
  int i = blockIdx.x * blockDim.x + threadIdx.x;
  if (i >= n) return;
  float a = x[i * 3], b = x[i * 3 + 1], c = x[i * 3 + 2];
  y[i * 3 + 0] = a * T[0] + b * T[3] + c * T[6];
  y[i * 3 + 1] = a * T[1] + b * T[4] + c * T[7];
  y[i * 3 + 2] = a * T[2] + b * T[5] + c * T[8];
}

// GroupNorm over point-major x[N,C]; stats per group over (Cg chans x N pts).
// Optionally mirrors the normalized result to f16 (feeds the next WMMA GEMM).
__global__ __launch_bounds__(256)
void k_gn_nc(float* __restrict__ x, const float* __restrict__ gw,
             const float* __restrict__ gb, int N, int C, int groups, int relu,
             _Float16* __restrict__ mir, int ldm) {
  __shared__ float sh[256];
  int g = blockIdx.x, Cg = C / groups;
  long cnt = (long)N * Cg;
  float s = 0.f, s2 = 0.f;
  for (long i = threadIdx.x; i < cnt; i += 256) {
    int p = (int)(i / Cg), c = g * Cg + (int)(i % Cg);
    float v = x[(size_t)p * C + c];
    s += v; s2 += v * v;
  }
  float S  = blockReduceSum256(s, sh);
  float S2 = blockReduceSum256(s2, sh);
  float mean = S / (float)cnt;
  float var  = S2 / (float)cnt - mean * mean;
  float rs   = rsqrtf(var + EPSN);
  for (long i = threadIdx.x; i < cnt; i += 256) {
    int p = (int)(i / Cg), c = g * Cg + (int)(i % Cg);
    size_t off = (size_t)p * C + c;
    float v = (x[off] - mean) * rs * gw[c] + gb[c];
    if (relu) v = fmaxf(v, 0.f);
    x[off] = v;
    if (mir) mir[(size_t)p * ldm + c] = (_Float16)v;
  }
}

// BatchNorm2d training stats per (segment g, channel o) over S*K samples.
__global__ __launch_bounds__(256)
void k_bn_seg(float* __restrict__ x, const float* __restrict__ gw,
              const float* __restrict__ gb, int SK, int O, int relu,
              _Float16* __restrict__ mir, int ldm) {
  __shared__ float sh[256];
  int o = blockIdx.x, g = blockIdx.y;
  size_t base = (size_t)g * SK * O + o;
  float s = 0.f, s2 = 0.f;
  for (int i = threadIdx.x; i < SK; i += 256) {
    float v = x[base + (size_t)i * O];
    s += v; s2 += v * v;
  }
  float S  = blockReduceSum256(s, sh);
  float S2 = blockReduceSum256(s2, sh);
  float mean = S / (float)SK;
  float var  = S2 / (float)SK - mean * mean;
  float rs   = rsqrtf(var + EPSN);
  for (int i = threadIdx.x; i < SK; i += 256) {
    size_t off = base + (size_t)i * O;
    float v = (x[off] - mean) * rs * gw[o] + gb[o];
    if (relu) v = fmaxf(v, 0.f);
    x[off] = v;
    if (mir) mir[((size_t)g * SK + i) * ldm + o] = (_Float16)v;
  }
}

__global__ __launch_bounds__(256)
void k_colmax(const float* __restrict__ x, float* __restrict__ y, int N, int C) {
  __shared__ float sh[256];
  int c = blockIdx.x;
  float m = -3.4e38f;
  for (int i = threadIdx.x; i < N; i += 256) m = fmaxf(m, x[(size_t)i * C + c]);
  sh[threadIdx.x] = m; __syncthreads();
  for (int off = 128; off > 0; off >>= 1) {
    if (threadIdx.x < off) sh[threadIdx.x] = fmaxf(sh[threadIdx.x], sh[threadIdx.x + off]);
    __syncthreads();
  }
  if (threadIdx.x == 0) y[c] = sh[0];
}

__global__ __launch_bounds__(256)
void k_fc(const float* __restrict__ x, const float* __restrict__ W,
          const float* __restrict__ b, float* __restrict__ y, int K) {
  __shared__ float sh[256];
  int o = blockIdx.x;
  float s = 0.f;
  for (int k = threadIdx.x; k < K; k += 256) s += x[k] * W[(size_t)o * K + k];
  float S = blockReduceSum256(s, sh);
  if (threadIdx.x == 0) y[o] = S + b[o];
}

__global__ __launch_bounds__(64)
void k_gn_vec(float* __restrict__ x, const float* __restrict__ gw,
              const float* __restrict__ gb, int C, int groups) {  // + ReLU
  __shared__ float sh[64];
  int g = blockIdx.x, Cg = C / groups, t = threadIdx.x;
  float s = 0.f, s2 = 0.f;
  for (int i = t; i < Cg; i += 64) { float v = x[g * Cg + i]; s += v; s2 += v * v; }
  sh[t] = s; __syncthreads();
  for (int o = 32; o > 0; o >>= 1) { if (t < o) sh[t] += sh[t + o]; __syncthreads(); }
  float S = sh[0]; __syncthreads();
  sh[t] = s2; __syncthreads();
  for (int o = 32; o > 0; o >>= 1) { if (t < o) sh[t] += sh[t + o]; __syncthreads(); }
  float S2 = sh[0]; __syncthreads();
  float mean = S / (float)Cg, var = S2 / (float)Cg - mean * mean;
  float rs = rsqrtf(var + EPSN);
  for (int i = t; i < Cg; i += 64) {
    int c = g * Cg + i;
    float v = (x[c] - mean) * rs * gw[c] + gb[c];
    x[c] = fmaxf(v, 0.f);
  }
}

__global__ __launch_bounds__(256)
void k_fc_eye(const float* __restrict__ x, const float* __restrict__ W,
              const float* __restrict__ b, float* __restrict__ T,
              float* __restrict__ Tout, int cout, int K) {
  __shared__ float sh[256];
  int o = blockIdx.x;
  float s = 0.f;
  for (int k = threadIdx.x; k < K; k += 256) s += x[k] * W[(size_t)o * K + k];
  float S = blockReduceSum256(s, sh);
  if (threadIdx.x == 0) {
    int i = o / cout, j = o % cout;
    float v = S + b[o] + ((i == j) ? 1.f : 0.f);
    T[o] = v;
    if (Tout) Tout[o] = v;
  }
}

// Iterative farthest-point sampling, one block per segment (n <= 1024).
// Segment xyz staged in LDS via gfx1250 async LDS loads when available.
__global__ __launch_bounds__(256)
void k_fps(const float* __restrict__ xyz, int* __restrict__ cidx, int n, int S) {
  __shared__ float sxyz[3 * 1024];
  __shared__ float dist[1024];
  __shared__ float rb[256];
  __shared__ int   ri[256];
  int g = blockIdx.x;
  const float* bx = xyz + (size_t)g * n * 3;
  const int tot = 3 * n;
#if HAVE_ASYNC_LDS
  for (int i = threadIdx.x; i < tot; i += 256)
    __builtin_amdgcn_global_load_async_to_lds_b32(AS1I(bx + i), AS3I(&sxyz[i]), 0, 0);
  __builtin_amdgcn_s_wait_asynccnt(0);
#else
  for (int i = threadIdx.x; i < tot; i += 256) sxyz[i] = bx[i];
#endif
  for (int i = threadIdx.x; i < n; i += 256) dist[i] = 1e10f;
  __syncthreads();
  int far = 0;
  for (int it = 0; it < S; ++it) {
    if (threadIdx.x == 0) cidx[(size_t)g * S + it] = far;
    float fx = sxyz[3 * far], fy = sxyz[3 * far + 1], fz = sxyz[3 * far + 2];
    float best = -1.f; int bi = 0;
    for (int i = threadIdx.x; i < n; i += 256) {
      float dx = sxyz[3 * i] - fx, dy = sxyz[3 * i + 1] - fy, dz = sxyz[3 * i + 2] - fz;
      float dd = fminf(dist[i], dx * dx + dy * dy + dz * dz);
      dist[i] = dd;
      if (dd > best) { best = dd; bi = i; }
    }
    __syncthreads();
    rb[threadIdx.x] = best; ri[threadIdx.x] = bi;
    __syncthreads();
    for (int off = 128; off > 0; off >>= 1) {   // argmax, first index on ties
      if (threadIdx.x < off) {
        float ob = rb[threadIdx.x + off]; int oi = ri[threadIdx.x + off];
        if (ob > rb[threadIdx.x] || (ob == rb[threadIdx.x] && oi < ri[threadIdx.x])) {
          rb[threadIdx.x] = ob; ri[threadIdx.x] = oi;
        }
      }
      __syncthreads();
    }
    far = ri[0];
    __syncthreads();
  }
}

// Ball query + grouping: index-ordered scan == reference's sort+truncate.
// One segment per block (blockIdx.y); segment xyz staged in LDS (async).
// Writes grouped [gxyz(3), pts(D)] straight to the f16 GEMM operand (padded).
__global__ __launch_bounds__(128)
void k_ballgroup(const float* __restrict__ xyz, const float* __restrict__ pts,
                 const int* __restrict__ cidx, float* __restrict__ new_xyz,
                 _Float16* __restrict__ A16,
                 int n, int D, int S, int ns, float r2, int Kp) {
  __shared__ float sxyz[3 * 1024];
  const int g = blockIdx.y;
  const float* bx = xyz + (size_t)g * n * 3;
  const int tot = 3 * n;
#if HAVE_ASYNC_LDS
  for (int i = threadIdx.x; i < tot; i += 128)
    __builtin_amdgcn_global_load_async_to_lds_b32(AS1I(bx + i), AS3I(&sxyz[i]), 0, 0);
  __builtin_amdgcn_s_wait_asynccnt(0);
#else
  for (int i = threadIdx.x; i < tot; i += 128) sxyz[i] = bx[i];
#endif
  __syncthreads();
  const int s = blockIdx.x * 128 + threadIdx.x;
  if (s >= S) return;
  const int t = g * S + s;
  const float* bp = pts + (size_t)g * n * D;
  int ci = cidx[t];
  float cx = sxyz[3 * ci], cy = sxyz[3 * ci + 1], cz = sxyz[3 * ci + 2];
  new_xyz[(size_t)t * 3 + 0] = cx;
  new_xyz[(size_t)t * 3 + 1] = cy;
  new_xyz[(size_t)t * 3 + 2] = cz;
  _Float16* dst = A16 + (size_t)t * ns * Kp;
  int cnt = 0, first = ci;
  for (int i = 0; i < n && cnt < ns; ++i) {
    float dx = sxyz[3 * i] - cx, dy = sxyz[3 * i + 1] - cy, dz = sxyz[3 * i + 2] - cz;
    if (dx * dx + dy * dy + dz * dz <= r2) {
      if (cnt == 0) first = i;
      _Float16* row = dst + (size_t)cnt * Kp;
      row[0] = (_Float16)dx; row[1] = (_Float16)dy; row[2] = (_Float16)dz;
      const float* pr = bp + (size_t)i * D;
      for (int c = 0; c < D; ++c) row[3 + c] = (_Float16)pr[c];
      ++cnt;
    }
  }
  if (cnt < ns) {   // pad with first in-radius hit (center always qualifies)
    float dx = sxyz[3 * first] - cx, dy = sxyz[3 * first + 1] - cy,
          dz = sxyz[3 * first + 2] - cz;
    const float* pr = bp + (size_t)first * D;
    for (; cnt < ns; ++cnt) {
      _Float16* row = dst + (size_t)cnt * Kp;
      row[0] = (_Float16)dx; row[1] = (_Float16)dy; row[2] = (_Float16)dz;
      for (int c = 0; c < D; ++c) row[3 + c] = (_Float16)pr[c];
    }
  }
}

__global__ void k_maxpool(const float* __restrict__ x, float* __restrict__ y,
                          long GS, int ns, int O) {
  long t = (long)blockIdx.x * blockDim.x + threadIdx.x;
  if (t >= GS * O) return;
  long gs = t / O; int o = (int)(t % O);
  const float* p = x + gs * (size_t)ns * O + o;
  float m = p[0];
  for (int k = 1; k < ns; ++k) m = fmaxf(m, p[(size_t)k * O]);
  y[t] = m;
}

__global__ void k_segmean(const float* __restrict__ h, float* __restrict__ f,
                          int S, int C) {
  int t = blockIdx.x * blockDim.x + threadIdx.x;
  if (t >= GSEG * C) return;
  int g = t / C, c = t % C;
  float s = 0.f;
  for (int i = 0; i < S; ++i) s += h[((size_t)g * S + i) * C + c];
  f[t] = s / (float)S;
}

__global__ void k_copy_f32(const float* __restrict__ s, float* __restrict__ d, int n) {
  int i = blockIdx.x * blockDim.x + threadIdx.x;
  if (i < n) d[i] = s[i];
}

// ------------------------------------------------------------- host helpers
static inline void launch_pad16(hipStream_t st, const float* src, _Float16* dst,
                                int rows, int cols, int colsP, int rowsP,
                                int trans, int srcld) {
  long tot = (long)rowsP * colsP;
  k_pad16<<<(int)((tot + 255) / 256), 256, 0, st>>>(src, dst, rows, cols, colsP,
                                                    tot, trans, srcld);
}
static inline void cvt_w(hipStream_t st, const float* W, _Float16* W16, int O,
                         int Cin, int Kp) {
  launch_pad16(st, W, W16, O, Cin, Kp, (O + 15) & ~15, 0, Cin);
}
static inline void launch_gemm(hipStream_t st, const _Float16* A,
                               const _Float16* B16, float* C, const float* bias,
                               int M, int N, int Kp, int relu) {
  dim3 grid((M / 16 + 7) / 8, (N + 63) / 64);
  k_wmma_gemm<<<grid, 256, 0, st>>>(A, B16, C, bias, M, N, Kp, N, relu);
}
static inline void launch_fill16(hipStream_t st, _Float16* p, long n) {
  k_fill_h16<<<4096, 256, 0, st>>>(p, n);
}

struct StnP {
  const float *w1, *b1, *g1w, *g1b, *w2, *b2, *g2w, *g2b, *w3, *b3, *g3w, *g3b,
      *f1w, *f1b, *fg1w, *fg1b, *f2w, *f2b, *fg2w, *fg2b, *ow, *ob;
};
static StnP stn_at(void* const* d, int b) {
  auto F = [&](int i) { return (const float*)d[b + i]; };
  StnP p{F(0),  F(1),  F(2),  F(3),  F(4),  F(5),  F(6),  F(7),
         F(8),  F(9),  F(10), F(11), F(12), F(13), F(14), F(15),
         F(16), F(17), F(18), F(19), F(20), F(21)};
  return p;
}
struct SaP { const float *W[3], *b[3], *gw[3], *gb[3]; };
static SaP sa_at(void* const* d, int base) {
  SaP s;
  for (int i = 0; i < 3; ++i) {
    s.W[i]  = (const float*)d[base + i];
    s.b[i]  = (const float*)d[base + 3 + i];
    s.gw[i] = (const float*)d[base + 6 + i];
    s.gb[i] = (const float*)d[base + 9 + i];
  }
  return s;
}

static void stn_forward(hipStream_t st, const StnP& p, const float* xNC, int N,
                        int cin, int cout, _Float16* A16, _Float16* W16,
                        float* X, float* v1024, float* u512, float* u256,
                        float* Tws, float* Tout) {
  const int K0 = (cin + 31) & ~31;
  launch_pad16(st, xNC, A16, N, cin, K0, N, 0, cin);
  cvt_w(st, p.w1, W16, 64, cin, K0);
  launch_gemm(st, A16, W16, X, p.b1, N, 64, K0, 0);
  k_gn_nc<<<64, 256, 0, st>>>(X, p.g1w, p.g1b, N, 64, 64, 1, A16, 64);
  cvt_w(st, p.w2, W16, 128, 64, 64);
  launch_gemm(st, A16, W16, X, p.b2, N, 128, 64, 0);
  k_gn_nc<<<128, 256, 0, st>>>(X, p.g2w, p.g2b, N, 128, 128, 1, A16, 128);
  cvt_w(st, p.w3, W16, 1024, 128, 128);
  launch_gemm(st, A16, W16, X, p.b3, N, 1024, 128, 0);
  k_gn_nc<<<1024, 256, 0, st>>>(X, p.g3w, p.g3b, N, 1024, 1024, 1, nullptr, 0);
  k_colmax<<<1024, 256, 0, st>>>(X, v1024, N, 1024);
  k_fc<<<512, 256, 0, st>>>(v1024, p.f1w, p.f1b, u512, 1024);
  k_gn_vec<<<512, 64, 0, st>>>(u512, p.fg1w, p.fg1b, 512, 512);
  k_fc<<<256, 256, 0, st>>>(u512, p.f2w, p.f2b, u256, 512);
  k_gn_vec<<<256, 64, 0, st>>>(u256, p.fg2w, p.fg2b, 256, 256);
  k_fc_eye<<<cout * cout, 256, 0, st>>>(u256, p.ow, p.ob, Tws, Tout, cout, 256);
}

static void sa_level(hipStream_t st, const SaP& p, const float* xyz,
                     const float* pts, int n, int D, int S, int ns, float radius,
                     int O0, int O1, int O2, int* cidx, float* new_xyz,
                     float* pooled, _Float16* H16, _Float16* W16, float* XA,
                     float* XB) {
  const int Kp0 = (3 + D + 31) & ~31;
  const int M   = GSEG * S * ns;   // multiple of 16 (G=32)
  const int SK  = S * ns;
  k_fps<<<GSEG, 256, 0, st>>>(xyz, cidx, n, S);
  launch_fill16(st, H16, (long)M * Kp0);            // zero channel padding
  k_ballgroup<<<dim3((S + 127) / 128, GSEG), 128, 0, st>>>(
      xyz, pts, cidx, new_xyz, H16, n, D, S, ns, radius * radius, Kp0);
  cvt_w(st, p.W[0], W16, O0, 3 + D, Kp0);
  launch_gemm(st, H16, W16, XA, p.b[0], M, O0, Kp0, 0);
  k_bn_seg<<<dim3(O0, GSEG), 256, 0, st>>>(XA, p.gw[0], p.gb[0], SK, O0, 1, H16, O0);
  cvt_w(st, p.W[1], W16, O1, O0, O0);
  launch_gemm(st, H16, W16, XB, p.b[1], M, O1, O0, 0);
  k_bn_seg<<<dim3(O1, GSEG), 256, 0, st>>>(XB, p.gw[1], p.gb[1], SK, O1, 1, H16, O1);
  cvt_w(st, p.W[2], W16, O2, O1, O1);
  launch_gemm(st, H16, W16, XA, p.b[2], M, O2, O1, 0);
  k_bn_seg<<<dim3(O2, GSEG), 256, 0, st>>>(XA, p.gw[2], p.gb[2], SK, O2, 1, nullptr, 0);
  long tot = (long)GSEG * S * O2;
  k_maxpool<<<(int)((tot + 255) / 256), 256, 0, st>>>(XA, pooled, (long)GSEG * S, ns, O2);
}

// ------------------------------------------------------------- entry point
// Input flattening (setup_inputs insertion order, nested dicts depth-first):
//   0: points | 1..22: stn1 | 23..44: stn2 | 45..56: sa1 | 57..68: sa2
//   69..80: sa3 | 81..88: head (w1,b1,g1w,g1b,w2,b2,g2w,g2b) | 89: point_split
extern "C" void kernel_launch(void* const* d_in, const int* in_sizes, int n_in,
                              void* d_out, int out_size, void* d_ws, size_t ws_size,
                              hipStream_t stream) {
  (void)in_sizes; (void)n_in; (void)out_size; (void)ws_size;
  const int S1 = 514, NS1 = 33, S2 = 259, NS2 = 23, S3 = 131, NS3 = 17;
  const int M1 = GSEG * S1, M3 = GSEG * S3;

  const float* points = (const float*)d_in[0];
  StnP stn1 = stn_at(d_in, 1);
  StnP stn2 = stn_at(d_in, 23);
  SaP  sa1  = sa_at(d_in, 45);
  SaP  sa2  = sa_at(d_in, 57);
  SaP  sa3  = sa_at(d_in, 69);
  const float *hw1 = (const float*)d_in[81], *hb1 = (const float*)d_in[82];
  const float *hg1w = (const float*)d_in[83], *hg1b = (const float*)d_in[84];
  const float *hw2 = (const float*)d_in[85], *hb2 = (const float*)d_in[86];
  const float *hg2w = (const float*)d_in[87], *hg2b = (const float*)d_in[88];

  float* out = (float*)d_out;          // [32,512] ++ trans1[9] ++ trans2[16384]
  float* outT1 = out + GSEG * 512;
  float* outT2 = outT1 + 9;

  // workspace bump allocator (harness-provided scratch)
  size_t off = 0;
  auto alloc = [&](size_t bytes) {
    void* r = (char*)d_ws + off;
    off += (bytes + 255) & ~(size_t)255;
    return r;
  };
  float*    XA   = (float*)alloc((size_t)71264 * 1024 * 4); // big fp32 ping
  float*    XB   = (float*)alloc((size_t)71264 * 512 * 4);  // big fp32 pong
  _Float16* H16  = (_Float16*)alloc((size_t)80 << 20);      // f16 activations
  _Float16* W16  = (_Float16*)alloc((size_t)4 << 20);       // f16 weights
  float* xT     = (float*)alloc((size_t)NT * 3 * 4);
  float* xyz0   = (float*)alloc((size_t)NT * 3 * 4);
  int*   cidx   = (int*)alloc((size_t)GSEG * S1 * 4);
  float* l1xyz  = (float*)alloc((size_t)M1 * 3 * 4);
  float* l1pool = (float*)alloc((size_t)M1 * 128 * 4);
  float* l1pT   = (float*)alloc((size_t)M1 * 128 * 4);
  float* l2xyz  = (float*)alloc((size_t)GSEG * S2 * 3 * 4);
  float* l2pool = (float*)alloc((size_t)GSEG * S2 * 256 * 4);
  float* l3xyz  = (float*)alloc((size_t)M3 * 3 * 4);
  float* l3pool = (float*)alloc((size_t)M3 * 1024 * 4);
  float* feats  = (float*)alloc((size_t)GSEG * 512 * 4);
  _Float16* f16feats = (_Float16*)alloc((size_t)GSEG * 512 * 2);
  float* out2   = (float*)alloc((size_t)GSEG * 512 * 4);
  float* v1024  = (float*)alloc(1024 * 4);
  float* u512   = (float*)alloc(512 * 4);
  float* u256   = (float*)alloc(256 * 4);
  float* T1ws   = (float*)alloc(16 * 4);
  float* T2ws   = (float*)alloc(16384 * 4);

  // 1) points [3,N] -> point-major [N,3]
  k_pts_transpose<<<(NT + 255) / 256, 256, 0, stream>>>(points, xT, NT);
  // 2) STN1 (3x3) and apply
  stn_forward(stream, stn1, xT, NT, 3, 3, H16, W16, XA, v1024, u512, u256,
              T1ws, outT1);
  k_apply3<<<(NT + 255) / 256, 256, 0, stream>>>(xT, T1ws, xyz0, NT);
  // 3) SA level 1: xyz=pts=xyz0 [32,1024,3]
  sa_level(stream, sa1, xyz0, xyz0, 1024, 3, S1, NS1, 0.4f, 64, 64, 128, cidx,
           l1xyz, l1pool, H16, W16, XA, XB);
  // 4) STN2 (128x128) on l1 features, then apply via WMMA GEMM
  stn_forward(stream, stn2, l1pool, M1, 128, 128, H16, W16, XA, v1024, u512,
              u256, T2ws, outT2);
  launch_pad16(stream, l1pool, H16, M1, 128, 128, M1, 0, 128);
  launch_pad16(stream, T2ws, W16, 128, 128, 128, 128, 1, 128);  // B = T2^T
  launch_gemm(stream, H16, W16, l1pT, nullptr, M1, 128, 128, 0);
  // 5) SA level 2 / 3
  sa_level(stream, sa2, l1xyz, l1pT, S1, 128, S2, NS2, 0.8f, 128, 128, 256,
           cidx, l2xyz, l2pool, H16, W16, XA, XB);
  sa_level(stream, sa3, l2xyz, l2pool, S2, 256, S3, NS3, 1.2f, 256, 512, 1024,
           cidx, l3xyz, l3pool, H16, W16, XA, XB);
  // 6) head: conv 1024->512 + GN(512) + ReLU, per-segment mean, conv 512->512
  launch_pad16(stream, l3pool, H16, M3, 1024, 1024, M3, 0, 1024);
  cvt_w(stream, hw1, W16, 512, 1024, 1024);
  launch_gemm(stream, H16, W16, XB, hb1, M3, 512, 1024, 0);
  k_gn_nc<<<512, 256, 0, stream>>>(XB, hg1w, hg1b, M3, 512, 512, 1, nullptr, 0);
  k_segmean<<<(GSEG * 512 + 255) / 256, 256, 0, stream>>>(XB, feats, S3, 512);
  launch_pad16(stream, feats, f16feats, GSEG, 512, 512, GSEG, 0, 512);
  cvt_w(stream, hw2, W16, 512, 512, 512);
  launch_gemm(stream, f16feats, W16, out2, hb2, GSEG, 512, 512, 0);
  k_gn_nc<<<16, 256, 0, stream>>>(out2, hg2w, hg2b, GSEG, 512, 16, 1, nullptr, 0);
  k_copy_f32<<<(GSEG * 512 + 255) / 256, 256, 0, stream>>>(out2, out, GSEG * 512);
}
